// KNNAttention_7773890806492
// MI455X (gfx1250) — matrix operand order repaired and checked by reading, not compiled, not run on hardware
//
#include <hip/hip_runtime.h>
#include <math.h>

// ---------------------------------------------------------------------------
// CDNA5 (gfx1250) wave32 WMMA / TDM types
// ---------------------------------------------------------------------------
typedef __attribute__((ext_vector_type(16))) __bf16   v16bf;
typedef __attribute__((ext_vector_type(8)))  float    v8f;
typedef __attribute__((ext_vector_type(4)))  unsigned v4u;
typedef __attribute__((ext_vector_type(8)))  int      v8i;
typedef __attribute__((ext_vector_type(4)))  int      v4i;

union BfFrag {
    v16bf    v;
    __bf16   e[16];
    unsigned u[8];
};

static __device__ __forceinline__ v8f wmma_bf16(const BfFrag& a, const BfFrag& b, v8f c) {
    return __builtin_amdgcn_wmma_f32_16x16x32_bf16(
        false, a.v, false, b.v, (short)0, c, false, false);
}

// Low 32 bits of a generic LDS pointer are the LDS byte offset.
static __device__ __forceinline__ unsigned lds_addr_of(const void* p) {
    return (unsigned)(size_t)p;
}

// ---------------------------------------------------------------------------
// Tensor Data Mover: 2D tile of 2-byte elements, global -> LDS.
//   tensor_d0/d1 : remaining extent from tile start (OOB rows/cols zero-fill)
//   stride0      : elements between dim1 rows in global memory
//   tile_d0      : contiguous elements per row, tile_d1: rows
// D# layout per CDNA5 ISA (group0) / §8.4 (group1).
// This toolchain exposes the 6-arg builtin (extra int32x8 group, unused here).
// Issue from ONE wave; TDM ignores EXEC. Tracked by TENSORcnt.
// ---------------------------------------------------------------------------
static __device__ __forceinline__ void tdm_load_2d_bf16(
    unsigned ldsAddr, const __bf16* gtile,
    unsigned tensor_d0, unsigned tensor_d1, long stride0,
    unsigned tile_d0, unsigned tile_d1)
{
    unsigned long long ga = (unsigned long long)(size_t)gtile;
    v4u g0;
    g0[0] = 1u;                                             // count=1, user D#
    g0[1] = ldsAddr;                                        // lds_addr [63:32]
    g0[2] = (unsigned)ga;                                   // global_addr lo
    g0[3] = (unsigned)((ga >> 32) & 0x1FFFFFFu) | (2u << 30); // addr hi | type=2
    unsigned long long s0 = (unsigned long long)stride0;
    v8i g1;
    g1[0] = (int)(1u << 16);                                // data_size=1 (2B)
    g1[1] = (int)((tensor_d0 & 0xFFFFu) << 16);             // tensor_dim0 lo16
    g1[2] = (int)((tensor_d0 >> 16) | ((tensor_d1 & 0xFFFFu) << 16));
    g1[3] = (int)((tensor_d1 >> 16) | (tile_d0 << 16));     // tile_dim0 @127:112
    g1[4] = (int)(tile_d1 & 0xFFFFu);                       // tile_dim1, tile_dim2=0
    g1[5] = (int)(unsigned)s0;                              // dim0_stride lo32
    g1[6] = (int)(unsigned)((s0 >> 32) & 0xFFFFu);          // dim0_stride hi16
    g1[7] = 0;
    v4i z4 = {0, 0, 0, 0};
    v8i z8 = {0, 0, 0, 0, 0, 0, 0, 0};
    __builtin_amdgcn_tensor_load_to_lds(g0, g1, z4, z4, z8, 0);
}

// ---------------------------------------------------------------------------
// One-time operand conversions
// ---------------------------------------------------------------------------
__global__ __launch_bounds__(256) void cvt_f32_bf16(
    const float* __restrict__ in, __bf16* __restrict__ out, long n)
{
    long stride = (long)gridDim.x * blockDim.x;
    for (long i = (long)blockIdx.x * blockDim.x + threadIdx.x; i < n; i += stride)
        out[i] = (__bf16)in[i];
}

// in: (K=1024, N) row-major fp32  ->  out: (N, K=1024) row-major bf16
__global__ __launch_bounds__(256) void transpose_f32_bf16_k1024(
    const float* __restrict__ in, __bf16* __restrict__ out, int N)
{
    long total  = (long)N << 10;
    long stride = (long)gridDim.x * blockDim.x;
    for (long i = (long)blockIdx.x * blockDim.x + threadIdx.x; i < total; i += stride) {
        int n = (int)(i >> 10), k = (int)(i & 1023);
        out[i] = (__bf16)in[(long)k * N + n];
    }
}

// ---------------------------------------------------------------------------
// Batched GEMM: C[b] = A[b](M,K) @ B[b](N,K)^T, all bf16 operands.
// Non-gather path stages tiles with the TDM; gather path (A rows indexed)
// stages manually. 128 threads / 4 waves; tile 64x64, K-step 32.
// ---------------------------------------------------------------------------
#define GKS 32

template <bool GATHER, bool COUT_F32>
__global__ __launch_bounds__(128) void gemm_bf16_wmma(
    const __bf16* __restrict__ A, long strideA, int lda,
    const __bf16* __restrict__ B, long strideB, int ldb,
    void* __restrict__ C, long strideC, int ldc,
    int M, int N, int K,
    const int* __restrict__ gatherIdx, int gatherStride)
{
    __shared__ __bf16 sA[64 * GKS];  // [m][k]
    __shared__ __bf16 sB[64 * GKS];  // [n][k]

    const int bz = blockIdx.z;
    const __bf16* Ab = A + (long)bz * strideA;
    const __bf16* Bb = B + (long)bz * strideB;
    const int* gidx = GATHER ? (gatherIdx + (long)bz * gatherStride) : nullptr;

    const int m0   = blockIdx.y * 64;
    const int n0   = blockIdx.x * 64;
    const int tid  = threadIdx.x;
    const int lane = tid & 31;
    const int wave = tid >> 5;
    const int wm   = (wave >> 1) * 32;
    const int wn   = (wave & 1) * 32;
    const int half = lane >> 4;
    const int l16  = lane & 15;

    v8f acc[2][2] = {};

    for (int k0 = 0; k0 < K; k0 += GKS) {
        if (!GATHER) {
            if (tid < 32) {  // wave 0 issues both DMA tiles (TDM ignores EXEC)
                tdm_load_2d_bf16(lds_addr_of(sA), Ab + (long)m0 * lda + k0,
                                 (unsigned)(K - k0), (unsigned)(M - m0), lda, GKS, 64);
                tdm_load_2d_bf16(lds_addr_of(sB), Bb + (long)n0 * ldb + k0,
                                 (unsigned)(K - k0), (unsigned)(N - n0), ldb, GKS, 64);
                __builtin_amdgcn_s_wait_tensorcnt(0);
            }
        } else {
            for (int e = tid; e < 64 * GKS; e += 128) {
                int r = e >> 5, c = e & 31;
                int gm = m0 + r;
                __bf16 val = (__bf16)0.f;
                if (gm < M) val = Ab[(long)gidx[gm] * lda + k0 + c];
                sA[e] = val;
            }
            for (int e = tid; e < 64 * GKS; e += 128) {
                int n = e >> 5, c = e & 31;
                int gn = n0 + n;
                sB[e] = (gn < N) ? Bb[(long)gn * ldb + k0 + c] : (__bf16)0.f;
            }
        }
        __syncthreads();

        BfFrag afr[2], bfr[2];
#pragma unroll
        for (int tm = 0; tm < 2; ++tm) {
            const __bf16* base = sA + (wm + tm * 16 + l16) * GKS + half * 16;
#pragma unroll
            for (int i = 0; i < 4; ++i) afr[tm].u[i]     = *(const unsigned*)(base + 2 * i);
#pragma unroll
            for (int i = 0; i < 4; ++i) afr[tm].u[4 + i] = *(const unsigned*)(base + 8 + 2 * i);
        }
#pragma unroll
        for (int tn = 0; tn < 2; ++tn) {
            const __bf16* base = sB + (wn + tn * 16 + l16) * GKS + half * 16;
#pragma unroll
            for (int i = 0; i < 8; ++i) bfr[tn].u[i] = *(const unsigned*)(base + 2 * i);
        }
#pragma unroll
        for (int tm = 0; tm < 2; ++tm)
#pragma unroll
            for (int tn = 0; tn < 2; ++tn)
                acc[tm][tn] = wmma_bf16(afr[tm], bfr[tn], acc[tm][tn]);
        __syncthreads();
    }

#pragma unroll
    for (int tm = 0; tm < 2; ++tm)
#pragma unroll
        for (int tn = 0; tn < 2; ++tn)
#pragma unroll
            for (int r = 0; r < 8; ++r) {
                int gm = m0 + wm + tm * 16 + half * 8 + r;
                int gn = n0 + wn + tn * 16 + l16;
                if (gm < M && gn < N) {
                    if (COUT_F32)
                        ((float*)C + (long)bz * strideC)[(long)gm * ldc + gn] = acc[tm][tn][r];
                    else
                        ((__bf16*)C + (long)bz * strideC)[(long)gm * ldc + gn] = (__bf16)acc[tm][tn][r];
                }
            }
}

// ---------------------------------------------------------------------------
// Row argmax (one wave per row), first-max tie-break like jnp.argmax
// ---------------------------------------------------------------------------
__global__ __launch_bounds__(256) void argmax_rows(
    const float* __restrict__ scores, int* __restrict__ idx, int rows, int n)
{
    int row  = blockIdx.x * 8 + (threadIdx.x >> 5);
    int lane = threadIdx.x & 31;
    if (row >= rows) return;
    const float* s = scores + (long)row * n;
    float best = -INFINITY;
    int   bi   = 0x7fffffff;
    for (int i = lane; i < n; i += 32) {
        float v = s[i];
        if (v > best) { best = v; bi = i; }
    }
#pragma unroll
    for (int m = 16; m >= 1; m >>= 1) {
        float ob = __shfl_xor(best, m, 32);
        int   oi = __shfl_xor(bi, m, 32);
        if (ob > best || (ob == best && oi < bi)) { best = ob; bi = oi; }
    }
    if (lane == 0) idx[row] = bi;
}

// ---------------------------------------------------------------------------
// Flash attention: per (b, h, 64-row block); K/V shared across the 16 heads.
// bf16 in, bf16 out; K tile staged by TDM, V staged manually (LDS transpose).
// ---------------------------------------------------------------------------
__global__ __launch_bounds__(128) void flash_attn(
    const __bf16* __restrict__ qp,    // (b, L, D)
    const __bf16* __restrict__ kvb,   // (b, L, 128): [0:64]=k, [64:128]=v
    __bf16* __restrict__ attn_out)    // (b, L, D)
{
    const int L = 2048, D = 1024, DH = 64;
    const int rb   = blockIdx.x;
    const int h    = blockIdx.y;
    const int b    = blockIdx.z;
    const int tid  = threadIdx.x;
    const int lane = tid & 31;
    const int wave = tid >> 5;
    const int half = lane >> 4;
    const int l16  = lane & 15;

    __shared__ __bf16 sK[64 * 64];     // [key][dh]
    __shared__ __bf16 sV[64 * 64];     // [dh][key]
    __shared__ __bf16 sP[4][16 * 64];  // per-wave P tile [m][key]

    const int mrow = rb * 64 + wave * 16 + l16;
    const __bf16* qrow = qp + ((long)b * L + mrow) * D + h * DH;
    BfFrag qf[2];
#pragma unroll
    for (int f = 0; f < 2; ++f)
#pragma unroll
        for (int e = 0; e < 16; ++e) {
            int k = f * 32 + half * 16 + e;
            qf[f].e[e] = qrow[k];
        }

    float m_i[8], l_i[8];
#pragma unroll
    for (int r = 0; r < 8; ++r) { m_i[r] = -INFINITY; l_i[r] = 0.f; }
    v8f o_acc[4] = {};

    for (int jb = 0; jb < L / 64; ++jb) {
        const __bf16* kvblk = kvb + ((long)b * L + jb * 64) * 128;
        if (tid < 32) {  // TDM: 64 keys x 64 dh, rows stride 128 elements
            tdm_load_2d_bf16(lds_addr_of(sK), kvblk,
                             128u, (unsigned)(L - jb * 64), 128, 64, 64);
            __builtin_amdgcn_s_wait_tensorcnt(0);
        }
        for (int e = tid; e < 64 * 64; e += 128) {
            int key = e >> 6, c = e & 63;
            sV[c * 64 + key] = kvblk[key * 128 + 64 + c];
        }
        __syncthreads();

        // S = (Q @ K^T) * 1/sqrt(dh)
        v8f s[4];
#pragma unroll
        for (int t = 0; t < 4; ++t) {
            BfFrag b0, b1;
            const __bf16* kb0 = sK + (t * 16 + l16) * 64 + half * 16;
#pragma unroll
            for (int i = 0; i < 8; ++i) {
                b0.u[i] = *(const unsigned*)(kb0 + 2 * i);
                b1.u[i] = *(const unsigned*)(kb0 + 32 + 2 * i);
            }
            v8f z = {};
            z = wmma_bf16(qf[0], b0, z);
            z = wmma_bf16(qf[1], b1, z);
            s[t] = z * 0.125f;
        }

        // online softmax (row r lives in VGPR r of the owning 16-lane half)
        float rmax[8];
#pragma unroll
        for (int r = 0; r < 8; ++r) {
            float v = fmaxf(fmaxf(s[0][r], s[1][r]), fmaxf(s[2][r], s[3][r]));
#pragma unroll
            for (int msk = 8; msk >= 1; msk >>= 1) v = fmaxf(v, __shfl_xor(v, msk, 32));
            rmax[r] = v;
        }
        float alpha[8], rsum[8];
#pragma unroll
        for (int r = 0; r < 8; ++r) {
            float mn = fmaxf(m_i[r], rmax[r]);
            alpha[r] = __expf(m_i[r] - mn);
            m_i[r]   = mn;
            rsum[r]  = 0.f;
        }
#pragma unroll
        for (int t = 0; t < 4; ++t)
#pragma unroll
            for (int r = 0; r < 8; ++r) {
                float p = __expf(s[t][r] - m_i[r]);
                rsum[r] += p;
                sP[wave][(half * 8 + r) * 64 + t * 16 + l16] = (__bf16)p;
            }
#pragma unroll
        for (int r = 0; r < 8; ++r) {
            float v = rsum[r];
#pragma unroll
            for (int msk = 8; msk >= 1; msk >>= 1) v += __shfl_xor(v, msk, 32);
            l_i[r] = l_i[r] * alpha[r] + v;
        }
#pragma unroll
        for (int t = 0; t < 4; ++t)
#pragma unroll
            for (int r = 0; r < 8; ++r) o_acc[t][r] *= alpha[r];

        // O += P @ V
        BfFrag pf[2];
        const __bf16* pbase = sP[wave] + l16 * 64;
#pragma unroll
        for (int f = 0; f < 2; ++f) {
            const __bf16* pb = pbase + f * 32 + half * 16;
#pragma unroll
            for (int i = 0; i < 4; ++i) pf[f].u[i]     = *(const unsigned*)(pb + 2 * i);
#pragma unroll
            for (int i = 0; i < 4; ++i) pf[f].u[4 + i] = *(const unsigned*)(pb + 8 + 2 * i);
        }
#pragma unroll
        for (int t = 0; t < 4; ++t) {
            BfFrag v0, v1;
            const __bf16* vb0 = sV + (t * 16 + l16) * 64 + half * 16;
#pragma unroll
            for (int i = 0; i < 8; ++i) {
                v0.u[i] = *(const unsigned*)(vb0 + 2 * i);
                v1.u[i] = *(const unsigned*)(vb0 + 32 + 2 * i);
            }
            o_acc[t] = wmma_bf16(pf[0], v0, o_acc[t]);
            o_acc[t] = wmma_bf16(pf[1], v1, o_acc[t]);
        }
        __syncthreads();
    }

#pragma unroll
    for (int t = 0; t < 4; ++t)
#pragma unroll
        for (int r = 0; r < 8; ++r) {
            int row = rb * 64 + wave * 16 + half * 8 + r;
            int col = h * DH + t * 16 + l16;
            attn_out[((long)b * L + row) * D + col] = (__bf16)(o_acc[t][r] / l_i[r]);
        }
}

// ---------------------------------------------------------------------------
// Orchestration
// ---------------------------------------------------------------------------
extern "C" void kernel_launch(void* const* d_in, const int* in_sizes, int n_in,
                              void* d_out, int out_size, void* d_ws, size_t ws_size,
                              hipStream_t stream) {
    const int B = 4, L = 2048, D = 1024, NM = 1000, H = 16, DH = 64;

    const float* q        = (const float*)d_in[0];
    const float* mem      = (const float*)d_in[2];
    const float* w_q      = (const float*)d_in[3];
    const float* w_kv     = (const float*)d_in[4];
    const float* w_concat = (const float*)d_in[5];

    char* wp = (char*)d_ws;
    auto carve = [&](size_t bytes) { char* p = wp; wp += (bytes + 255) & ~(size_t)255; return p; };
    __bf16* qbf    = (__bf16*)carve((size_t)B * L * D * 2);
    __bf16* membf  = (__bf16*)carve((size_t)B * NM * D * 2);
    __bf16* wqT    = (__bf16*)carve((size_t)D * D * 2);
    __bf16* wkvT   = (__bf16*)carve((size_t)2 * DH * D * 2);
    __bf16* wcT    = (__bf16*)carve((size_t)D * D * 2);
    __bf16* qp     = (__bf16*)carve((size_t)B * L * D * 2);
    __bf16* attnb  = (__bf16*)carve((size_t)B * L * D * 2);
    __bf16* kvbuf  = (__bf16*)carve((size_t)B * L * 2 * DH * 2);
    float*  scores = (float*)carve((size_t)B * L * NM * 4);
    int*    idx    = (int*)carve((size_t)B * L * 4);

    dim3 blk128(128);

    // 0) one-time bf16 conversions; weights transposed to (N, K)
    cvt_f32_bf16<<<2048, 256, 0, stream>>>(q, qbf, (long)B * L * D);
    cvt_f32_bf16<<<1024, 256, 0, stream>>>(mem, membf, (long)B * NM * D);
    transpose_f32_bf16_k1024<<<512, 256, 0, stream>>>(w_q, wqT, D);
    transpose_f32_bf16_k1024<<<64, 256, 0, stream>>>(w_kv, wkvT, 2 * DH);
    transpose_f32_bf16_k1024<<<512, 256, 0, stream>>>(w_concat, wcT, D);

    // 1) qp = q @ w_q                (bf16 out)
    gemm_bf16_wmma<false, false><<<dim3(D / 64, L / 64, B), blk128, 0, stream>>>(
        qbf, (long)L * D, D, wqT, 0, D, qp, (long)L * D, D, L, D, D, nullptr, 0);

    // 2) scores = qp @ mem_table^T   (f32 out for argmax)
    gemm_bf16_wmma<false, true><<<dim3((NM + 63) / 64, L / 64, B), blk128, 0, stream>>>(
        qp, (long)L * D, D, membf, (long)NM * D, D, scores, (long)L * NM, NM,
        L, NM, D, nullptr, 0);

    // 3) idx = argmax(scores)
    argmax_rows<<<(B * L) / 8, 256, 0, stream>>>(scores, idx, B * L, NM);

    // 4) kv = mem_table[idx] @ w_kv  (gather path, bf16 out)
    gemm_bf16_wmma<true, false><<<dim3(2 * DH / 64, L / 64, B), blk128, 0, stream>>>(
        membf, (long)NM * D, D, wkvT, 0, D, kvbuf, (long)L * 2 * DH, 2 * DH,
        L, 2 * DH, D, idx, L);

    // 5) flash attention -> attnb (bf16)
    flash_attn<<<dim3(L / 64, H, B), blk128, 0, stream>>>(qp, kvbuf, attnb);

    // 6) out = attn @ w_concat       (f32 out -> d_out)
    gemm_bf16_wmma<false, true><<<dim3(D / 64, L / 64, B), blk128, 0, stream>>>(
        attnb, (long)L * D, D, wcT, 0, D, (float*)d_out, (long)L * D, D,
        L, D, D, nullptr, 0);
}
